// ConformerAttention_21947282882809
// MI455X (gfx1250) — matrix operand order repaired and checked by reading, not compiled, not run on hardware
//
#include <hip/hip_runtime.h>
#include <hip/hip_bf16.h>
#include <cstdint>

// ConformerAttention for MI455X (gfx1250, wave32, WMMA f16 -> f32 acc).
// Workspace requirement: ~287 MB (xn16/q/k/v/ao16 f16 buffers + 115 MB pos bias).

typedef _Float16 half_t;
typedef __attribute__((ext_vector_type(16))) _Float16 v16h;
typedef __attribute__((ext_vector_type(8)))  _Float16 v8h;
typedef __attribute__((ext_vector_type(8)))  float    v8f;
typedef __attribute__((ext_vector_type(4)))  unsigned int u32x4;
typedef __attribute__((ext_vector_type(8)))  unsigned int u32x8;

#define DIMN  512
#define NHEAD 8
#define DH    64
#define CTX   200
#define RP    224                  // CTX padded to 14*16 for WMMA tiling
#define BATCH 4
#define TLEN  8000
#define NBLK  40                   // TLEN / CTX
#define BMH   (BATCH * NBLK * NHEAD)   // 1280
#define NROWS (BATCH * TLEN)           // 32000
#define ATT_SCALE 0.125f               // DH^-0.5

// ---------------------------------------------------------------- fragments

__device__ __forceinline__ v8f zero8f() {
  v8f z;
#pragma unroll
  for (int i = 0; i < 8; ++i) z[i] = 0.f;
  return z;
}

__device__ __forceinline__ v16h zero16h() {
  v16h z;
#pragma unroll
  for (int i = 0; i < 16; ++i) z[i] = (_Float16)0.f;
  return z;
}

// A fragment (16xK, K=32 chunk). Memory row-major [row][k], leading dim ld.
// Lane L: row = L&15; K base = (L>>4)*16; 16 contiguous K values per lane
// (matches load_bt).  (not the ISA table k order; valid since load_bt uses the same)
__device__ __forceinline__ v16h load_a(const half_t* __restrict__ base, int row0,
                                       int ld, int k0, int lane, int rows_valid) {
  int m   = lane & 15;
  int off = (lane >> 4) * 16;
  if (m >= rows_valid) return zero16h();
  const half_t* p = base + (size_t)(row0 + m) * ld + k0 + off;
  v8h lo = *(const v8h*)p;
  v8h hi = *(const v8h*)(p + 8);
  v16h a;
#pragma unroll
  for (int j = 0; j < 8; ++j) { a[j] = lo[j]; a[j + 8] = hi[j]; }
  return a;
}

// B fragment (KxN, K=32 chunk) where memory holds B^T row-major: mem[n][k], ld.
// Lane L: col n = L&15; K base = (L>>4)*16; 16 contiguous K values per lane.
__device__ __forceinline__ v16h load_bt(const half_t* __restrict__ base, int col0,
                                        int ld, int k0, int lane) {
  int n  = lane & 15;
  int kb = (lane >> 4) * 16;
  const half_t* p = base + (size_t)(col0 + n) * ld + k0 + kb;
  v8h lo = *(const v8h*)p;
  v8h hi = *(const v8h*)(p + 8);
  v16h b;
#pragma unroll
  for (int j = 0; j < 8; ++j) { b[j] = lo[j]; b[j + 8] = hi[j]; }
  return b;
}

__device__ __forceinline__ v8f wmma16(v16h a, v16h b, v8f c) {
  return __builtin_amdgcn_wmma_f32_16x16x32_f16(false, a, false, b, (short)0, c,
                                                false, false);
}

// ---------------------------------------------------------------- kernels

// LayerNorm, one wave32 per 512-element row, output f16.
__global__ void k_layernorm(const float* __restrict__ x, const float* __restrict__ g,
                            const float* __restrict__ beta, half_t* __restrict__ xn) {
  int lane = threadIdx.x & 31;
  int row  = blockIdx.x * (blockDim.x >> 5) + (threadIdx.x >> 5);
  if (row >= NROWS) return;
  const float* xr = x + (size_t)row * DIMN;
  float v[16];
  float s = 0.f, s2 = 0.f;
#pragma unroll
  for (int i = 0; i < 16; ++i) {
    v[i] = xr[lane + i * 32];
    s += v[i];
    s2 += v[i] * v[i];
  }
#pragma unroll
  for (int m = 1; m < 32; m <<= 1) {
    s  += __shfl_xor(s, m, 32);
    s2 += __shfl_xor(s2, m, 32);
  }
  float mu  = s * (1.f / DIMN);
  float var = s2 * (1.f / DIMN) - mu * mu;
  float r   = rsqrtf(var + 1e-5f);
  half_t* o = xn + (size_t)row * DIMN;
#pragma unroll
  for (int i = 0; i < 16; ++i) {
    int c = lane + i * 32;
    o[c] = (half_t)((v[i] - mu) * r * g[c] + beta[c]);
  }
}

// Transpose+convert weights to f16: wcatT = [Wq|Wkv]^T (1536 x 512),
// woT = Wo^T (512 x 512), so GEMM B-fragments become contiguous per lane.
__global__ void k_prep_w(const float* __restrict__ Wq, const float* __restrict__ Wkv,
                         const float* __restrict__ Wo, half_t* __restrict__ wcatT,
                         half_t* __restrict__ woT) {
  int idx = blockIdx.x * blockDim.x + threadIdx.x;
  if (idx < 1536 * DIMN) {
    int n = idx / DIMN, k = idx % DIMN;          // wcatT[n][k]
    float v = (n < DIMN) ? Wq[(size_t)k * DIMN + n]
                         : Wkv[(size_t)k * 1024 + (n - DIMN)];
    wcatT[idx] = (half_t)v;
  }
  if (idx < DIMN * DIMN) {
    int n = idx / DIMN, k = idx % DIMN;          // woT[n][k]
    woT[idx] = (half_t)Wo[(size_t)k * DIMN + n];
  }
}

// rel16[c][r][d] = pos_emb[attention_dists[c][r]][d], r padded to 224 with zeros.
__global__ void k_prep_rel(const int* __restrict__ dists, const float* __restrict__ pe,
                           half_t* __restrict__ rel16) {
  int idx = blockIdx.x * blockDim.x + threadIdx.x;
  if (idx >= CTX * RP * DH) return;
  int d = idx & 63;
  int r = (idx >> 6) % RP;
  int c = idx / (RP * DH);
  float v = 0.f;
  if (r < CTX) v = pe[(size_t)dists[c * CTX + r] * DH + d];
  rel16[idx] = (half_t)v;
}

// Fused QKV GEMM: xn16 (32000x512) @ [Wq|Wkv] (512x1536) via wcatT, WMMA,
// scatter into blocked [bmh][c][d] layouts. 4 waves/block, 16x64 strip each.
__global__ void k_qkv_gemm(const half_t* __restrict__ xn, const half_t* __restrict__ wcatT,
                           half_t* __restrict__ qb, half_t* __restrict__ kb,
                           half_t* __restrict__ vb) {
  int lane  = threadIdx.x & 31;
  int wave  = threadIdx.x >> 5;
  int strip = blockIdx.x;               // 0..23 (64 output cols each)
  int row0  = (blockIdx.y * 4 + wave) * 16;
  v8f acc[4];
#pragma unroll
  for (int dt = 0; dt < 4; ++dt) acc[dt] = zero8f();
  for (int k0 = 0; k0 < DIMN; k0 += 32) {
    v16h a = load_a(xn, row0, DIMN, k0, lane, 16);
#pragma unroll
    for (int dt = 0; dt < 4; ++dt) {
      v16h b = load_bt(wcatT, strip * 64 + dt * 16, DIMN, k0, lane);
      acc[dt] = wmma16(a, b, acc[dt]);
    }
  }
  int n = lane & 15, mb = (lane >> 4) * 8;
  // row -> (b, block, c) decomposition once per lane, then increment w/ wrap
  int rowbase = row0 + mb;                    // 8-aligned; TLEN,CTX boundaries ok
  int bi  = rowbase / TLEN;
  int t0  = rowbase - bi * TLEN;
  int mblk0 = t0 / CTX;
  int c0    = t0 - mblk0 * CTX;
#pragma unroll
  for (int dt = 0; dt < 4; ++dt) {
    int colbase = strip * 64 + dt * 16;       // wave-uniform
    half_t* dst;
    int cc;
    if (colbase < DIMN)          { dst = qb; cc = colbase; }
    else if (colbase < 2 * DIMN) { dst = kb; cc = colbase - DIMN; }
    else                         { dst = vb; cc = colbase - 2 * DIMN; }
    int h = cc >> 6, d = (cc & 63) + n;
#pragma unroll
    for (int i = 0; i < 8; ++i) {
      int c = c0 + i, mblk = mblk0;
      if (c >= CTX) { c -= CTX; mblk += 1; }  // at most one wrap within 8 rows
      int bmh = (bi * NBLK + mblk) * NHEAD + h;
      dst[((size_t)bmh * CTX + c) * DH + d] = (half_t)acc[dt][i];
    }
  }
}

// pos bias: for fixed query position c, GEMM (1280 x 64) @ rel[c]^T (64 x 224),
// scaled by DH^-0.5, stored f16 as pos16[bmh][c][r].
__global__ void k_pos_gemm(const half_t* __restrict__ qb, const half_t* __restrict__ rel16,
                           half_t* __restrict__ pos16) {
  extern __shared__ char smem[];
  half_t* relLds = (half_t*)smem;       // RP*DH halves = 28672 B
  int c   = blockIdx.x;
  int tid = threadIdx.x;
  const half_t* src = rel16 + (size_t)c * RP * DH;
  for (int idx = tid; idx < RP * DH / 8; idx += blockDim.x)
    *(v8h*)(relLds + idx * 8) = *(const v8h*)(src + idx * 8);
  __syncthreads();

  int lane  = tid & 31, wave = tid >> 5;
  int mtile = blockIdx.y * 4 + wave;    // 0..79 (1280 rows / 16)
  const half_t* qbase = qb + (size_t)c * DH;   // row stride CTX*DH over bmh
  v16h a0 = load_a(qbase, mtile * 16, CTX * DH, 0, lane, 16);
  v16h a1 = load_a(qbase, mtile * 16, CTX * DH, 32, lane, 16);
  int n = lane & 15, mb = (lane >> 4) * 8;
  for (int nt = 0; nt < RP / 16; ++nt) {
    v8f acc = zero8f();
    v16h b0 = load_bt(relLds, nt * 16, DH, 0, lane);
    v16h b1 = load_bt(relLds, nt * 16, DH, 32, lane);
    acc = wmma16(a0, b0, acc);
    acc = wmma16(a1, b1, acc);
#pragma unroll
    for (int i = 0; i < 8; ++i) {
      int row = mtile * 16 + mb + i;
      pos16[((size_t)row * CTX + c) * RP + nt * 16 + n] =
          (half_t)(acc[i] * ATT_SCALE);
    }
  }
}

// Attention per (b, block, head): K tile DMA'd into LDS by the Tensor Data
// Mover, V transposed into LDS by VALU, register softmax with wave32 shuffles,
// probabilities re-laid-out through per-wave LDS for the PV WMMAs.
__global__ void __launch_bounds__(256) k_attention(
    const half_t* __restrict__ qb, const half_t* __restrict__ kb,
    const half_t* __restrict__ vb, const half_t* __restrict__ pos16,
    half_t* __restrict__ ao16) {
  extern __shared__ char smem[];
  half_t* kLds    = (half_t*)smem;          // [r][d] : RP*DH
  half_t* vLds    = kLds + RP * DH;         // [d][r] : DH*RP (transposed)
  half_t* attnLds = vLds + RP * DH;         // 8 waves * 16*RP
  int bmh  = blockIdx.x;
  int tid  = threadIdx.x, lane = tid & 31, wave = tid >> 5;
  const half_t* kg = kb + (size_t)bmh * CTX * DH;
  const half_t* vg = vb + (size_t)bmh * CTX * DH;

  // --- TDM: async-load the 200x64 f16 K tile into LDS (wave 0 issues; TDM
  // ignores EXEC so exactly one wave must execute the instruction).
  if (tid < 32) {
    unsigned long long ga = (unsigned long long)(uintptr_t)kg;
    u32x4 g0;
    g0[0] = 1u;                                   // count=1, no gather
    g0[1] = (unsigned int)(uintptr_t)kLds;        // lds_addr (aperture low bits)
    g0[2] = (unsigned int)ga;                     // global_addr[31:0]
    g0[3] = (unsigned int)((ga >> 32) & 0x01FFFFFFu) | (2u << 30);  // type=2
    u32x8 g1;
    g1[0] = 1u << 16;                             // data_size = 2 bytes
    g1[1] = (unsigned int)DH << 16;               // tensor_dim0 = 64 (lo16)
    g1[2] = (unsigned int)CTX << 16;              // tensor_dim1 = 200 (lo16)
    g1[3] = (unsigned int)DH << 16;               // tile_dim0 = 64
    g1[4] = (unsigned int)CTX;                    // tile_dim1 = 200, tile_dim2=0
    g1[5] = (unsigned int)DH;                     // tensor_dim0_stride = 64
    g1[6] = 0u;                                   // stride hi / dim1_stride
    g1[7] = 0u;
    asm volatile("tensor_load_to_lds %0, %1" :: "s"(g0), "s"(g1) : "memory");
  }
  // zero-pad K rows [200,224) while the TDM runs
  for (int idx = tid; idx < (RP - CTX) * DH; idx += blockDim.x)
    kLds[CTX * DH + idx] = (half_t)0.f;
  // V: load row-major, store transposed [d][r] (zero-padded r)
  for (int idx = tid; idx < RP * DH / 8; idx += blockDim.x) {
    int r = idx >> 3;
    int d0 = (idx & 7) * 8;
    v8h vv;
    if (r < CTX) {
      vv = *(const v8h*)(vg + (size_t)r * DH + d0);
    } else {
#pragma unroll
      for (int j = 0; j < 8; ++j) vv[j] = (_Float16)0.f;
    }
#pragma unroll
    for (int j = 0; j < 8; ++j) vLds[(size_t)(d0 + j) * RP + r] = vv[j];
  }
  if (tid < 32) __builtin_amdgcn_s_wait_tensorcnt(0);
  __syncthreads();

  const half_t* qbase = qb + (size_t)bmh * CTX * DH;
  half_t* attnW = attnLds + (size_t)wave * 16 * RP;
  int n = lane & 15, mb = (lane >> 4) * 8;
  int bi   = bmh / (NBLK * NHEAD);
  int rem  = bmh % (NBLK * NHEAD);
  int mblk = rem / NHEAD, h = rem % NHEAD;

  for (int ct = wave; ct < (CTX + 15) / 16; ct += 8) {   // 13 query tiles
    int rows_valid = min(16, CTX - ct * 16);
    v16h a0 = load_a(qbase, ct * 16, DH, 0, lane, rows_valid);
    v16h a1 = load_a(qbase, ct * 16, DH, 32, lane, rows_valid);
    if (lane < rows_valid) {           // hint the pos-bias rows into cache
      const half_t* pp = pos16 + ((size_t)bmh * CTX + ct * 16 + lane) * RP;
      __builtin_prefetch(pp, 0, 3);
      __builtin_prefetch(pp + 128, 0, 3);
    }
    v8f acc[RP / 16];
#pragma unroll
    for (int rt = 0; rt < RP / 16; ++rt) acc[rt] = zero8f();
#pragma unroll
    for (int rt = 0; rt < RP / 16; ++rt) {
      v16h b0 = load_bt(kLds, rt * 16, DH, 0, lane);
      v16h b1 = load_bt(kLds, rt * 16, DH, 32, lane);
      acc[rt] = wmma16(a0, b0, acc[rt]);
      acc[rt] = wmma16(a1, b1, acc[rt]);
    }
    // scale + relative-position bias + column mask
#pragma unroll
    for (int rt = 0; rt < RP / 16; ++rt) {
      int r = rt * 16 + n;
#pragma unroll
      for (int i = 0; i < 8; ++i) {
        int c = ct * 16 + mb + i;
        float s = acc[rt][i] * ATT_SCALE;
        if (c < CTX && r < CTX)
          s += (float)pos16[((size_t)bmh * CTX + c) * RP + r];
        if (r >= CTX) s = -1e30f;
        acc[rt][i] = s;
      }
    }
    // softmax over r (rows live in one VGPR across a 16-lane half)
    float mx[8], sum[8];
#pragma unroll
    for (int i = 0; i < 8; ++i) mx[i] = -1e30f;
#pragma unroll
    for (int rt = 0; rt < RP / 16; ++rt)
#pragma unroll
      for (int i = 0; i < 8; ++i) mx[i] = fmaxf(mx[i], acc[rt][i]);
#pragma unroll
    for (int i = 0; i < 8; ++i) {
#pragma unroll
      for (int m = 1; m < 16; m <<= 1) mx[i] = fmaxf(mx[i], __shfl_xor(mx[i], m, 32));
      sum[i] = 0.f;
    }
#pragma unroll
    for (int rt = 0; rt < RP / 16; ++rt)
#pragma unroll
      for (int i = 0; i < 8; ++i) {
        float e = __expf(acc[rt][i] - mx[i]);
        acc[rt][i] = e;
        sum[i] += e;
      }
#pragma unroll
    for (int i = 0; i < 8; ++i) {
#pragma unroll
      for (int m = 1; m < 16; m <<= 1) sum[i] += __shfl_xor(sum[i], m, 32);
      sum[i] = 1.f / sum[i];
    }
    // write f16 probabilities to this wave's LDS scratch (re-layout D -> A)
#pragma unroll
    for (int rt = 0; rt < RP / 16; ++rt)
#pragma unroll
      for (int i = 0; i < 8; ++i)
        attnW[(size_t)(mb + i) * RP + rt * 16 + n] = (half_t)(acc[rt][i] * sum[i]);
    asm volatile("s_wait_dscnt 0" ::: "memory");   // intra-wave DS RAW fence

    // out = P (16x224) @ V (224x64)
    v8f oacc[4];
#pragma unroll
    for (int dt = 0; dt < 4; ++dt) oacc[dt] = zero8f();
    for (int kc = 0; kc < RP; kc += 32) {
      v16h a = load_a(attnW, 0, RP, kc, lane, 16);
#pragma unroll
      for (int dt = 0; dt < 4; ++dt) {
        v16h b = load_bt(vLds, dt * 16, RP, kc, lane);  // vLds is V^T [d][r]
        oacc[dt] = wmma16(a, b, oacc[dt]);
      }
    }
#pragma unroll
    for (int dt = 0; dt < 4; ++dt)
#pragma unroll
      for (int i = 0; i < 8; ++i) {
        int c = ct * 16 + mb + i;
        if (c < CTX) {
          int t = mblk * CTX + c;
          ao16[(size_t)(bi * TLEN + t) * DIMN + h * DH + dt * 16 + n] =
              (half_t)oacc[dt][i];
        }
      }
  }
}

// Output GEMM: ao16 (32000x512) @ Wo (512x512, stored transposed) + bo -> f32.
__global__ void k_out_gemm(const half_t* __restrict__ ao16, const half_t* __restrict__ woT,
                           const float* __restrict__ bo, float* __restrict__ out) {
  int lane  = threadIdx.x & 31;
  int wave  = threadIdx.x >> 5;
  int strip = blockIdx.x;               // 0..7
  int row0  = (blockIdx.y * 4 + wave) * 16;
  v8f acc[4];
#pragma unroll
  for (int dt = 0; dt < 4; ++dt) acc[dt] = zero8f();
  for (int k0 = 0; k0 < DIMN; k0 += 32) {
    v16h a = load_a(ao16, row0, DIMN, k0, lane, 16);
#pragma unroll
    for (int dt = 0; dt < 4; ++dt) {
      v16h b = load_bt(woT, strip * 64 + dt * 16, DIMN, k0, lane);
      acc[dt] = wmma16(a, b, acc[dt]);
    }
  }
  int n = lane & 15, mb = (lane >> 4) * 8;
#pragma unroll
  for (int dt = 0; dt < 4; ++dt) {
    int col = strip * 64 + dt * 16 + n;
    float bias = bo[col];
#pragma unroll
    for (int i = 0; i < 8; ++i)
      out[(size_t)(row0 + mb + i) * DIMN + col] = acc[dt][i] + bias;
  }
}

// ---------------------------------------------------------------- launch

extern "C" void kernel_launch(void* const* d_in, const int* in_sizes, int n_in,
                              void* d_out, int out_size, void* d_ws, size_t ws_size,
                              hipStream_t stream) {
  (void)in_sizes; (void)n_in; (void)out_size; (void)ws_size;
  const float* x     = (const float*)d_in[0];
  const int*   dists = (const int*)d_in[1];
  const float* ln_g  = (const float*)d_in[2];
  const float* ln_b  = (const float*)d_in[3];
  const float* Wq    = (const float*)d_in[4];
  const float* Wkv   = (const float*)d_in[5];
  const float* pe    = (const float*)d_in[6];
  const float* Wo    = (const float*)d_in[7];
  const float* bo    = (const float*)d_in[8];
  float* out = (float*)d_out;

  char* ws = (char*)d_ws;
  size_t off = 0;
  auto take = [&](size_t bytes) -> char* {
    char* p = ws + off;
    off += (bytes + 255) & ~(size_t)255;
    return p;
  };
  half_t* xn16  = (half_t*)take((size_t)NROWS * DIMN * 2);          // 32.8 MB
  half_t* wcatT = (half_t*)take((size_t)DIMN * 1536 * 2);           //  1.6 MB
  half_t* woT   = (half_t*)take((size_t)DIMN * DIMN * 2);           //  0.5 MB
  half_t* q16   = (half_t*)take((size_t)BMH * CTX * DH * 2);        // 32.8 MB
  half_t* k16   = (half_t*)take((size_t)BMH * CTX * DH * 2);        // 32.8 MB
  half_t* v16b  = (half_t*)take((size_t)BMH * CTX * DH * 2);        // 32.8 MB
  half_t* rel16 = (half_t*)take((size_t)CTX * RP * DH * 2);         //  5.7 MB
  half_t* pos16 = (half_t*)take((size_t)BMH * CTX * RP * 2);        // 114.7 MB
  half_t* ao16  = (half_t*)take((size_t)NROWS * DIMN * 2);          // 32.8 MB

  k_layernorm<<<NROWS / 8, 256, 0, stream>>>(x, ln_g, ln_b, xn16);
  k_prep_w<<<(1536 * DIMN + 255) / 256, 256, 0, stream>>>(Wq, Wkv, Wo, wcatT, woT);
  k_prep_rel<<<(CTX * RP * DH + 255) / 256, 256, 0, stream>>>(dists, pe, rel16);
  k_qkv_gemm<<<dim3(24, NROWS / 64), 128, 0, stream>>>(xn16, wcatT, q16, k16, v16b);
  k_pos_gemm<<<dim3(CTX, 20), 128, RP * DH * 2, stream>>>(q16, rel16, pos16);
  k_attention<<<BMH, 256, (2 * RP * DH + 8 * 16 * RP) * 2, stream>>>(
      q16, k16, v16b, pos16, ao16);
  k_out_gemm<<<dim3(8, NROWS / 64), 128, 0, stream>>>(ao16, woT, bo, out);
}
